// ScaledDotProductAttention_25975962206788
// MI455X (gfx1250) — matrix operand order repaired and checked by reading, not compiled, not run on hardware
//
#include <hip/hip_runtime.h>
#include <hip/hip_bf16.h>

// Flash attention (causal) for B=1, H=16, S=4096, D=64, fp32 in/out.
// Pass 1: transpose+convert K,V to bf16 in workspace (fragment-major layouts).
// Pass 2: flash attention; all matrix math via v_wmma_f32_16x16x32_bf16.
//         32 query rows per wave, 32 keys per iteration, wave-wide running
//         softmax max (valid upper bound; bf16 P is scale-invariant in
//         relative precision), row-sums via WMMA against an all-ones B.
//         Exponentials use raw v_exp_f32 (flush-to-zero underflow is the
//         mathematically desired behavior here).

#define ATT_H 16
#define ATT_S 4096
#define ATT_D 64

// 1/sqrt(64) * log2(e): softmax runs in the exp2 domain.
#define QSCALE (0.125f * 1.4426950408889634f)

typedef __attribute__((ext_vector_type(16))) __bf16 v16bf;
typedef __attribute__((ext_vector_type(8)))  float  v8f;
typedef __attribute__((ext_vector_type(4)))  unsigned int uint4v;

union frag16 { v16bf bf; uint4v u[2]; };

__device__ __forceinline__ v16bf load_frag_32B(const __bf16* p) {
    frag16 f;
    f.u[0] = *(const uint4v*)(p);      // global_load_b128
    f.u[1] = *(const uint4v*)(p + 8);  // global_load_b128
    return f.bf;
}

__device__ __forceinline__ float wave32_max(float v) {
    v = fmaxf(v, __shfl_xor(v, 1, 32));
    v = fmaxf(v, __shfl_xor(v, 2, 32));
    v = fmaxf(v, __shfl_xor(v, 4, 32));
    v = fmaxf(v, __shfl_xor(v, 8, 32));
    v = fmaxf(v, __shfl_xor(v, 16, 32));
    return v;
}

// ---------------------------------------------------------------------------
// Pass 1: per-head transpose + f32->bf16 convert.
//   src: [Z][R][C] f32   ->   dst: [Z][C][R] bf16
// ---------------------------------------------------------------------------
__global__ __launch_bounds__(256)
void transpose_cvt_bf16(const float* __restrict__ src, __bf16* __restrict__ dst,
                        int R, int C) {
    __shared__ float tile[32][33];
    const int h  = blockIdx.z;
    const float* s = src + (size_t)h * R * C;
    __bf16*      d = dst + (size_t)h * R * C;
    const int c0 = blockIdx.x * 32;
    const int r0 = blockIdx.y * 32;
    const int tx = threadIdx.x;      // 0..31
    const int ty = threadIdx.y;      // 0..7
    #pragma unroll
    for (int i = 0; i < 32; i += 8)
        tile[ty + i][tx] = s[(size_t)(r0 + ty + i) * C + c0 + tx];
    __syncthreads();
    #pragma unroll
    for (int i = 0; i < 32; i += 8)
        d[(size_t)(c0 + ty + i) * R + r0 + tx] = (__bf16)tile[tx][ty + i];
}

// ---------------------------------------------------------------------------
// One 32-key tile: scores (8 WMMA), online softmax, P@V (8 WMMA) + rowsum (2).
// ---------------------------------------------------------------------------
template<bool MASKED>
__device__ __forceinline__ void fa_tile(
    int t0, int qb, int half, int ln,
    const __bf16* __restrict__ kbase, const __bf16* __restrict__ vbase,
    const v16bf (&qa)[2][2], const v16bf& ones,
    v8f (&acc)[2][4], float& mrun, float (&lstat)[2][8],
    __bf16 (&ldsPw)[2][16][32]) {

    // ---- K fragments: [s-tile][kt]; 16 consecutive bf16 per lane -> b128 x2.
    v16bf bk[2][2];
    #pragma unroll
    for (int s = 0; s < 2; ++s)
        #pragma unroll
        for (int kt = 0; kt < 2; ++kt)
            bk[s][kt] = load_frag_32B(kbase + (size_t)(t0 + 16 * s + ln) * ATT_D
                                            + 32 * kt + 16 * half);

    // ---- Scores: 2 M-tiles x 2 key sub-tiles, accumulate over D (8 WMMAs).
    v8f sc[2][2];
    #pragma unroll
    for (int mt = 0; mt < 2; ++mt)
        #pragma unroll
        for (int s = 0; s < 2; ++s) {
            v8f t = (v8f){0.f,0.f,0.f,0.f,0.f,0.f,0.f,0.f};
            #pragma unroll
            for (int kt = 0; kt < 2; ++kt)
                t = __builtin_amdgcn_wmma_f32_16x16x32_bf16(false, qa[mt][kt],
                        false, bk[s][kt], (short)0, t, false, false);
            sc[mt][s] = t;
        }

    if (MASKED) {  // reference semantics: masked entries become exactly -1e9
        #pragma unroll
        for (int mt = 0; mt < 2; ++mt)
            #pragma unroll
            for (int r = 0; r < 8; ++r) {
                const int qrow = qb + 16 * mt + r + 8 * half;
                if (t0 + ln      > qrow) sc[mt][0][r] = -1.0e9f;
                if (t0 + 16 + ln > qrow) sc[mt][1][r] = -1.0e9f;
            }
    }

    // ---- V fragments (shared by both M-tiles).
    v16bf bv[4];
    #pragma unroll
    for (int nt = 0; nt < 4; ++nt)
        bv[nt] = load_frag_32B(vbase + (size_t)(16 * nt + ln) * ATT_S
                                     + t0 + 16 * half);

    // ---- Wave-wide tile max (upper bound of every row max in this wave).
    float tm = sc[0][0][0];
    #pragma unroll
    for (int mt = 0; mt < 2; ++mt)
        #pragma unroll
        for (int s = 0; s < 2; ++s)
            #pragma unroll
            for (int r = 0; r < 8; ++r)
                tm = fmaxf(tm, sc[mt][s][r]);
    tm = wave32_max(tm);

    // ---- Rare, wave-uniform rescale: running max only grows.
    if (tm > mrun) {
        const float alpha = __builtin_amdgcn_exp2f(mrun - tm);  // raw v_exp_f32
        mrun = tm;
        #pragma unroll
        for (int mt = 0; mt < 2; ++mt) {
            #pragma unroll
            for (int nt = 0; nt < 4; ++nt)
                #pragma unroll
                for (int r = 0; r < 8; ++r)
                    acc[mt][nt][r] *= alpha;
            #pragma unroll
            for (int r = 0; r < 8; ++r)
                lstat[mt][r] *= alpha;
        }
    }

    #pragma unroll
    for (int mt = 0; mt < 2; ++mt) {
        // ---- P = exp2(S - m), staged to LDS (C-layout -> A-layout transpose).
        //      Raw v_exp_f32: underflow flushes to 0, which is exactly what a
        //      vanished softmax term (or a -1e9 masked entry) should produce.
        #pragma unroll
        for (int r = 0; r < 8; ++r) {
            ldsPw[mt][r + 8 * half][ln] =
                (__bf16)__builtin_amdgcn_exp2f(sc[mt][0][r] - mrun);
            ldsPw[mt][r + 8 * half][16 + ln] =
                (__bf16)__builtin_amdgcn_exp2f(sc[mt][1][r] - mrun);
        }
        asm volatile("" ::: "memory");  // DS ops are in-order per wave

        // ---- Rebuild P as an A-layout 16x32 bf16 fragment (b32 LDS reads).
        v16bf pa;
        unsigned int* pw = (unsigned int*)&pa;
        #pragma unroll
        for (int j = 0; j < 8; ++j) {
            const int kk = 16 * half + 2 * j;
            pw[j] = *(const unsigned int*)&ldsPw[mt][ln][kk];
        }

        // ---- Row sums via WMMA: P(16x32) @ ones(32x16); result layout matches
        //      lstat (uniform across each 16-lane half).
        v8f rs = __builtin_amdgcn_wmma_f32_16x16x32_bf16(false, pa, false, ones,
                     (short)0, (v8f){0.f,0.f,0.f,0.f,0.f,0.f,0.f,0.f}, false, false);
        #pragma unroll
        for (int r = 0; r < 8; ++r)
            lstat[mt][r] += rs[r];

        // ---- acc += P @ V (4 WMMAs per M-tile).
        #pragma unroll
        for (int nt = 0; nt < 4; ++nt)
            acc[mt][nt] = __builtin_amdgcn_wmma_f32_16x16x32_bf16(false, pa,
                              false, bv[nt], (short)0, acc[mt][nt], false, false);
    }
}

// ---------------------------------------------------------------------------
// Pass 2: causal flash attention.
//   Kb: [H][S][D] bf16  -> score B-frags contiguous per lane
//   Vb: [H][D][S] bf16  -> PV    B-frags contiguous per lane
// ---------------------------------------------------------------------------
__global__ __launch_bounds__(128)
void fa_causal_bf16_wmma(const float* __restrict__ q,
                         const __bf16* __restrict__ Kb,
                         const __bf16* __restrict__ Vb,
                         float* __restrict__ out) {
    __shared__ __bf16 ldsP[4][2][16][32];   // per-wave P staging

    const int lane = threadIdx.x & 31;
    const int wav  = threadIdx.x >> 5;
    const int half = lane >> 4;
    const int ln   = lane & 15;
    const int h    = blockIdx.y;
    const int qb   = blockIdx.x * 128 + wav * 32;   // 32 query rows per wave

    const float*  qbase = q  + ((size_t)h * ATT_S + qb) * ATT_D;
    const __bf16* kbase = Kb + (size_t)h * ATT_S * ATT_D;   // [t][d]
    const __bf16* vbase = Vb + (size_t)h * ATT_D * ATT_S;   // [d][t]

    // ---- Q fragments (A-layout 16x32 bf16): [mt][kt], scale folded in.
    v16bf qa[2][2];
    #pragma unroll
    for (int mt = 0; mt < 2; ++mt)
        #pragma unroll
        for (int kt = 0; kt < 2; ++kt)
            #pragma unroll
            for (int j = 0; j < 8; ++j) {
                const int kk = 16 * half + 2 * j;
                const int dd = 32 * kt + kk;
                const size_t row = (size_t)(16 * mt + ln) * ATT_D;
                qa[mt][kt][2 * j]     = (__bf16)(qbase[row + dd]     * QSCALE);
                qa[mt][kt][2 * j + 1] = (__bf16)(qbase[row + dd + 1] * QSCALE);
            }

    v16bf ones;   // all-ones B matrix for WMMA row sums
    {
        unsigned int* ow = (unsigned int*)&ones;
        #pragma unroll
        for (int j = 0; j < 8; ++j) ow[j] = 0x3F803F80u;   // bf16(1.0) x2
    }

    v8f acc[2][4];
    float lstat[2][8];
    float mrun = -3.0e38f;
    #pragma unroll
    for (int mt = 0; mt < 2; ++mt) {
        #pragma unroll
        for (int nt = 0; nt < 4; ++nt)
            acc[mt][nt] = (v8f){0.f,0.f,0.f,0.f,0.f,0.f,0.f,0.f};
        #pragma unroll
        for (int r = 0; r < 8; ++r) lstat[mt][r] = 0.0f;
    }

    // qb is a multiple of 32 -> exactly one masked (diagonal) tile, peeled.
    const int nfull = qb >> 5;
    for (int tt = 0; tt < nfull; ++tt) {
        const int t0 = tt * 32;
        __builtin_prefetch(kbase + (size_t)(t0 + 32) * ATT_D, 0, 1);  // next tile
        __builtin_prefetch(vbase + (t0 + 32), 0, 1);
        fa_tile<false>(t0, qb, half, ln, kbase, vbase, qa, ones,
                       acc, mrun, lstat, ldsP[wav]);
    }
    fa_tile<true>(qb, qb, half, ln, kbase, vbase, qa, ones,
                  acc, mrun, lstat, ldsP[wav]);

    // ---- Normalize and write out (C-layout -> [S, D] row-major).
    //      v_rcp_f32 relative error (~2^-23..2^-14 class) is far below the
    //      bf16 quantization already in the pipeline.
    float* obase = out + ((size_t)h * ATT_S + qb) * ATT_D;
    #pragma unroll
    for (int mt = 0; mt < 2; ++mt)
        #pragma unroll
        for (int r = 0; r < 8; ++r) {
            const float inv = __builtin_amdgcn_rcpf(lstat[mt][r]);
            const int mm = 16 * mt + r + 8 * half;
            #pragma unroll
            for (int nt = 0; nt < 4; ++nt)
                obase[(size_t)mm * ATT_D + nt * 16 + ln] = acc[mt][nt][r] * inv;
        }
}

extern "C" void kernel_launch(void* const* d_in, const int* in_sizes, int n_in,
                              void* d_out, int out_size, void* d_ws, size_t ws_size,
                              hipStream_t stream) {
    const float* q = (const float*)d_in[0];
    const float* k = (const float*)d_in[1];   // [H][D][S]
    const float* v = (const float*)d_in[2];   // [H][S][D]
    float* out = (float*)d_out;

    __bf16* Kb = (__bf16*)d_ws;                           // [H][S][D] bf16 (8 MB)
    __bf16* Vb = Kb + (size_t)ATT_H * ATT_S * ATT_D;      // [H][D][S] bf16 (8 MB)

    // K [H][D][S] -> Kb [H][S][D]   (R=D, C=S)
    transpose_cvt_bf16<<<dim3(ATT_S / 32, ATT_D / 32, ATT_H), dim3(32, 8), 0, stream>>>(
        k, Kb, ATT_D, ATT_S);
    // V [H][S][D] -> Vb [H][D][S]   (R=S, C=D)
    transpose_cvt_bf16<<<dim3(ATT_D / 32, ATT_S / 32, ATT_H), dim3(32, 8), 0, stream>>>(
        v, Vb, ATT_S, ATT_D);

    fa_causal_bf16_wmma<<<dim3(ATT_S / 128, ATT_H), dim3(128), 0, stream>>>(q, Kb, Vb, out);
}
